// Self_Dynamic_Prototype_52080773431770
// MI455X (gfx1250) — compile-verified, matrix operand order checked by reading, not run on hardware
//
#include <hip/hip_runtime.h>
#include <hip/hip_bf16.h>

// ---------------- problem constants ----------------
#define WAYN   5
#define SHOTN  5
#define PP     10
#define FD     640
#define HD     512
#define NH     8
#define DKH    64
#define S_HW   196
#define B_ALL  100
#define NQN    75
#define MB     80            // WAY + NQ batches through MHA
#define M1     19600         // B_ALL * S_HW
#define MM     15680         // MB * S_HW
#define MQ     14700         // NQ * S_HW
#define NSUP   980           // WAY * S_HW
#define OUT_UPD 9408000      // 75*640*196

typedef __attribute__((ext_vector_type(4)))  unsigned int u32x4;
typedef __attribute__((ext_vector_type(4)))  int          v4i;
typedef __attribute__((ext_vector_type(16))) __bf16       bf16x16;
typedef __attribute__((ext_vector_type(8)))  float        f32x8;

union Frag { bf16x16 v; u32x4 q[2]; unsigned short h[16]; };
static_assert(sizeof(Frag) == 32, "frag size");

// CDNA5 async global->LDS path (ASYNCcnt-tracked). Guarded: falls back to a
// register bounce if the builtins are absent on this toolchain.
#if defined(__gfx1250__) && __has_builtin(__builtin_amdgcn_global_load_async_to_lds_b128) && __has_builtin(__builtin_amdgcn_s_wait_asynccnt)
#define USE_ASYNC_LDS 1
#else
#define USE_ASYNC_LDS 0
#endif

__device__ inline unsigned short f32_to_bf16(float f) {
  unsigned int u = __float_as_uint(f);
  u += 0x7FFFu + ((u >> 16) & 1u);          // round-to-nearest-even
  return (unsigned short)(u >> 16);
}

__device__ inline f32x8 wmma_bf16(const Frag& a, const Frag& b, f32x8 c) {
  return __builtin_amdgcn_wmma_f32_16x16x32_bf16(
      /*neg_a=*/false, a.v, /*neg_b=*/false, b.v,
      /*c_mod=*/(short)0, c, /*reuse_a=*/false, /*reuse_b=*/false);
}

// ---------------- generic f32 -> bf16 convert ----------------
__global__ __launch_bounds__(256)
void f2bf_kernel(const float* __restrict__ in, unsigned short* __restrict__ out, int n) {
  int i = blockIdx.x * 256 + threadIdx.x;
  if (i < n) out[i] = f32_to_bf16(in[i]);
}

// pack key NCHW (n,c,hw) -> A1[(n*196+hw)*640 + c] bf16
__global__ __launch_bounds__(256)
void pack_key_kernel(const float* __restrict__ key, unsigned short* __restrict__ A1) {
  long long idx = (long long)blockIdx.x * 256 + threadIdx.x;
  if (idx >= (long long)M1 * FD) return;
  int m = (int)(idx / FD), c = (int)(idx % FD);
  int n = m / S_HW, hw = m % S_HW;
  A1[idx] = f32_to_bf16(key[((size_t)n * FD + c) * S_HW + hw]);
}

// ---------------- TN bf16 WMMA GEMM: C[m,n] = sum_k A[m,k]*W[n,k] (+bias[n]) ----------------
// Weight slab (64 rows x K) is staged once per block into LDS via the async
// global->LDS engine; all 4 waves then pull B fragments with ds_load_b128.
// MODE 0: f32 out + bias ; MODE 1: bf16 out + bias ; MODE 2: f32 out + bias + residual
template<int MODE>
__device__ inline void epi_store(const f32x8& c, int mt, int nt, int lane, int M, int N,
                                 const float* __restrict__ bias, const float* __restrict__ Res,
                                 float* __restrict__ Cf, unsigned short* __restrict__ Ch) {
  int col = nt + (lane & 15);
  int rb  = mt + (lane >> 4) * 8;
  float bv = bias[col];
#pragma unroll
  for (int e = 0; e < 8; e++) {
    int row = rb + e;
    if (row < M) {
      float v = c[e] + bv;
      if (MODE == 2) v += Res[(size_t)row * N + col];
      if (MODE == 1) Ch[(size_t)row * N + col] = f32_to_bf16(v);
      else           Cf[(size_t)row * N + col] = v;
    }
  }
}

template<int MODE>
__global__ __launch_bounds__(128)
void gemm_tn_wmma(const unsigned short* __restrict__ A,  // M x K bf16
                  const unsigned short* __restrict__ W,  // N x K bf16
                  const float* __restrict__ bias,        // N
                  const float* __restrict__ Res,         // M x N (MODE 2)
                  float* __restrict__ Cf,
                  unsigned short* __restrict__ Ch,
                  int M, int N, int K) {
  extern __shared__ unsigned short sW[];   // 64 x K bf16 weight slab

  // ---- cooperative stage of this block's weight slab into LDS ----
  {
    const unsigned short* src = W + (size_t)blockIdx.x * 64 * K;
    int chunks = (64 * K) >> 3;            // 16 bytes per chunk
    for (int i = threadIdx.x; i < chunks; i += 128) {
#if USE_ASYNC_LDS
      __builtin_amdgcn_global_load_async_to_lds_b128(
          (v4i*)(src + (size_t)i * 8),
          (v4i*)(sW + (size_t)i * 8), 0, 0);
#else
      *(u32x4*)(sW + (size_t)i * 8) = *(const u32x4*)(src + (size_t)i * 8);
#endif
    }
#if USE_ASYNC_LDS
    __builtin_amdgcn_s_wait_asynccnt(0);
#endif
    __syncthreads();
  }

  int wave = threadIdx.x >> 5, lane = threadIdx.x & 31;
  int wm = wave >> 1, wn = wave & 1;
  int m0 = blockIdx.y * 64 + wm * 32;
  int n0 = blockIdx.x * 64 + wn * 32;
  int lr = lane & 15, kh = lane >> 4;

  int ra0 = min(m0 + lr,      M - 1);
  int ra1 = min(m0 + 16 + lr, M - 1);
  const unsigned short* pa0 = A + (size_t)ra0 * K + kh * 8;
  const unsigned short* pa1 = A + (size_t)ra1 * K + kh * 8;
  const unsigned short* pb0 = sW + (size_t)(wn * 32 + lr) * K + kh * 8;       // LDS
  const unsigned short* pb1 = sW + (size_t)(wn * 32 + 16 + lr) * K + kh * 8;  // LDS

  f32x8 c00 = {0.f,0.f,0.f,0.f,0.f,0.f,0.f,0.f};
  f32x8 c01 = c00, c10 = c00, c11 = c00;

  for (int k = 0; k < K; k += 32) {
    Frag a0, a1, b0, b1;
    a0.q[0] = *(const u32x4*)(pa0 + k);  a0.q[1] = *(const u32x4*)(pa0 + k + 16);
    a1.q[0] = *(const u32x4*)(pa1 + k);  a1.q[1] = *(const u32x4*)(pa1 + k + 16);
    b0.q[0] = *(const u32x4*)(pb0 + k);  b0.q[1] = *(const u32x4*)(pb0 + k + 16);
    b1.q[0] = *(const u32x4*)(pb1 + k);  b1.q[1] = *(const u32x4*)(pb1 + k + 16);
    if (k + 32 < K) {
      __builtin_prefetch(pa0 + k + 32, 0, 0);
      __builtin_prefetch(pa1 + k + 32, 0, 0);
    }
    c00 = wmma_bf16(a0, b0, c00);
    c01 = wmma_bf16(a0, b1, c01);
    c10 = wmma_bf16(a1, b0, c10);
    c11 = wmma_bf16(a1, b1, c11);
  }
  epi_store<MODE>(c00, m0,      n0,      lane, M, N, bias, Res, Cf, Ch);
  epi_store<MODE>(c01, m0,      n0 + 16, lane, M, N, bias, Res, Cf, Ch);
  epi_store<MODE>(c10, m0 + 16, n0,      lane, M, N, bias, Res, Cf, Ch);
  epi_store<MODE>(c11, m0 + 16, n0 + 16, lane, M, N, bias, Res, Cf, Ch);
}

// ---------------- BN statistics (mean/var per column over M rows) ----------------
__global__ __launch_bounds__(256)
void bn_stats_kernel(const float* __restrict__ Y, int M, int N, float* __restrict__ stats) {
  int c = blockIdx.x;
  __shared__ float sh[256], sh2[256];
  float s = 0.f, s2 = 0.f;
  for (int m = threadIdx.x; m < M; m += 256) {
    float v = Y[(size_t)m * N + c];
    s += v; s2 += v * v;
  }
  sh[threadIdx.x] = s; sh2[threadIdx.x] = s2; __syncthreads();
  for (int o = 128; o > 0; o >>= 1) {
    if (threadIdx.x < o) { sh[threadIdx.x] += sh[threadIdx.x + o]; sh2[threadIdx.x] += sh2[threadIdx.x + o]; }
    __syncthreads();
  }
  if (threadIdx.x == 0) {
    float mu = sh[0] / (float)M;
    stats[c] = mu;
    stats[N + c] = sh2[0] / (float)M - mu * mu;
  }
}

// BN normalize + relu -> f32
__global__ __launch_bounds__(256)
void bn_relu_kernel(const float* __restrict__ Y, const float* __restrict__ stats,
                    const float* __restrict__ g, const float* __restrict__ bta,
                    float* __restrict__ out, int M, int N) {
  long long idx = (long long)blockIdx.x * 256 + threadIdx.x;
  if (idx >= (long long)M * N) return;
  int c = (int)(idx % N);
  float v = (Y[idx] - stats[c]) * rsqrtf(stats[N + c] + 1e-5f) * g[c] + bta[c];
  out[idx] = v > 0.f ? v : 0.f;
}

// build MHA input: rows 0..979 = support mean over shots, rows 980.. = query feats
__global__ __launch_bounds__(256)
void build_x_kernel(const float* __restrict__ feats, float* __restrict__ x,
                    unsigned short* __restrict__ xh) {
  long long idx = (long long)blockIdx.x * 256 + threadIdx.x;
  if (idx >= (long long)MM * HD) return;
  int row = (int)(idx / HD), d = (int)(idx % HD);
  float v;
  if (row < NSUP) {
    int w = row / S_HW, s = row % S_HW;
    v = 0.f;
#pragma unroll
    for (int sh = 0; sh < SHOTN; sh++)
      v += feats[(((size_t)(sh * WAYN + w)) * S_HW + s) * HD + d];
    v *= 0.2f;
  } else {
    int qrow = row - NSUP;
    v = feats[((size_t)(WAYN * SHOTN) * S_HW + qrow) * HD + d];
  }
  x[idx] = v;
  xh[idx] = f32_to_bf16(v);
}

// ---------------- attention: one block per (b,h), scores+softmax in LDS ----------------
#define ATT_SP   224
#define ATT_ROWS 208
#define ATT_ST   13
#define ATT_SMEM (ATT_ROWS * ATT_SP * 4 + ATT_ROWS * ATT_SP * 2)

__global__ __launch_bounds__(256)
void attn_kernel(const unsigned short* __restrict__ Qb,
                 const unsigned short* __restrict__ Kb,
                 const unsigned short* __restrict__ Vb,
                 float* __restrict__ Ctx) {
  extern __shared__ char smem[];
  float* sS = (float*)smem;                                            // 208 x 224 f32
  unsigned short* sA = (unsigned short*)(smem + ATT_ROWS * ATT_SP * 4); // 208 x 224 bf16
  int b = blockIdx.x >> 3, h = blockIdx.x & 7;
  int wave = threadIdx.x >> 5, lane = threadIdx.x & 31;
  int lr = lane & 15, kh = lane >> 4;
  size_t base = (size_t)b * S_HW * HD + (size_t)h * DKH;

  // phase 1: scores = (q k^T) / 8
  for (int t = wave; t < ATT_ST * ATT_ST; t += 8) {
    int ti = t / ATT_ST, tj = t % ATT_ST;
    int qi = min(ti * 16 + lr, S_HW - 1);
    int kj = min(tj * 16 + lr, S_HW - 1);
    const unsigned short* pq = Qb + base + (size_t)qi * HD + kh * 8;
    const unsigned short* pk = Kb + base + (size_t)kj * HD + kh * 8;
    f32x8 acc = {0.f,0.f,0.f,0.f,0.f,0.f,0.f,0.f};
#pragma unroll
    for (int k = 0; k < DKH; k += 32) {
      Frag a, bb;
      a.q[0]  = *(const u32x4*)(pq + k);  a.q[1]  = *(const u32x4*)(pq + k + 16);
      bb.q[0] = *(const u32x4*)(pk + k);  bb.q[1] = *(const u32x4*)(pk + k + 16);
      acc = wmma_bf16(a, bb, acc);
    }
#pragma unroll
    for (int e = 0; e < 8; e++) {
      int i = ti * 16 + e + kh * 8;
      int j = tj * 16 + lr;
      sS[i * ATT_SP + j] = acc[e] * 0.125f;
    }
  }
  __syncthreads();

  // phase 2: row softmax -> bf16 att (zero-padded columns)
  for (int i = threadIdx.x; i < S_HW; i += 256) {
    float mx = -3.4e38f;
    for (int j = 0; j < S_HW; j++) mx = fmaxf(mx, sS[i * ATT_SP + j]);
    float sum = 0.f;
    for (int j = 0; j < S_HW; j++) { float e = __expf(sS[i * ATT_SP + j] - mx); sS[i * ATT_SP + j] = e; sum += e; }
    float inv = 1.f / sum;
    for (int j = 0; j < S_HW; j++) sA[i * ATT_SP + j] = f32_to_bf16(sS[i * ATT_SP + j] * inv);
    for (int j = S_HW; j < ATT_SP; j++) sA[i * ATT_SP + j] = 0;
  }
  __syncthreads();

  // phase 3: ctx = att @ v  (A from LDS, B strided from global)
  for (int t = wave; t < ATT_ST * 4; t += 8) {
    int ti = t >> 2, td = t & 3;
    int ai = ti * 16 + lr;
    int dcol = td * 16 + lr;
    f32x8 acc = {0.f,0.f,0.f,0.f,0.f,0.f,0.f,0.f};
    for (int k = 0; k < ATT_SP; k += 32) {
      Frag a, bb;
      const unsigned short* pa = sA + ai * ATT_SP + k + kh * 8;
      a.q[0] = *(const u32x4*)(pa);
      a.q[1] = *(const u32x4*)(pa + 16);
#pragma unroll
      for (int e = 0; e < 8; e++) {
        int j0 = min(k + kh * 8 + e,      S_HW - 1);
        int j1 = min(k + 16 + kh * 8 + e, S_HW - 1);
        bb.h[e]     = Vb[base + (size_t)j0 * HD + dcol];
        bb.h[8 + e] = Vb[base + (size_t)j1 * HD + dcol];
      }
      acc = wmma_bf16(a, bb, acc);
    }
#pragma unroll
    for (int e = 0; e < 8; e++) {
      int i = ti * 16 + e + kh * 8;
      if (i < S_HW) Ctx[base + (size_t)i * HD + dcol] = acc[e];
    }
  }
}

// ---------------- layernorm in place (one wave per row) ----------------
__global__ __launch_bounds__(256)
void layernorm_kernel(float* __restrict__ X, const float* __restrict__ g,
                      const float* __restrict__ b, int M) {
  int row = blockIdx.x * 8 + (threadIdx.x >> 5);
  if (row >= M) return;
  int lane = threadIdx.x & 31;
  float* p = X + (size_t)row * HD;
  float s = 0.f, s2 = 0.f;
  for (int i = lane; i < HD; i += 32) { float v = p[i]; s += v; s2 += v * v; }
  for (int o = 16; o > 0; o >>= 1) { s += __shfl_xor(s, o, 32); s2 += __shfl_xor(s2, o, 32); }
  float mu = s / (float)HD;
  float inv = rsqrtf(s2 / (float)HD - mu * mu + 1e-5f);
  for (int i = lane; i < HD; i += 32) p[i] = (p[i] - mu) * inv * g[i] + b[i];
}

// ---------------- prototypes: logits, column softmax over 980, weighted sum ----------------
__global__ __launch_bounds__(256)
void proto_kernel(const float* __restrict__ sup, const float* __restrict__ Wm,
                  float* __restrict__ protos) {
  __shared__ float L[NSUP * PP];
  __shared__ float cmx[PP], csum[PP];
  int tid = threadIdx.x;
  for (int r = tid; r < NSUP; r += 256) {
    const float* row = sup + (size_t)r * HD;
    for (int p = 0; p < PP; p++) {
      float s = 0.f;
      for (int d = 0; d < HD; d++) s += row[d] * Wm[p * HD + d];
      L[r * PP + p] = s;
    }
  }
  __syncthreads();
  if (tid < PP) {
    float mx = -3.4e38f;
    for (int r = 0; r < NSUP; r++) mx = fmaxf(mx, L[r * PP + tid]);
    float s = 0.f;
    for (int r = 0; r < NSUP; r++) s += __expf(L[r * PP + tid] - mx);
    cmx[tid] = mx; csum[tid] = s;
  }
  __syncthreads();
  for (int r = tid; r < NSUP; r += 256)
    for (int p = 0; p < PP; p++)
      L[r * PP + p] = __expf(L[r * PP + p] - cmx[p]) / csum[p];
  __syncthreads();
  for (int pd = tid; pd < PP * HD; pd += 256) {
    int p = pd / HD, d = pd % HD;
    float s = 0.f;
    for (int r = 0; r < NSUP; r++) s += L[r * PP + p] * sup[(size_t)r * HD + d];
    protos[pd] = s;
  }
}

// l2norm protos + dis_loss + cst_loss
__global__ __launch_bounds__(256)
void proto_loss_kernel(const float* __restrict__ protos, float* __restrict__ pn,
                       float* __restrict__ outsc) {
  __shared__ float nrm[PP];
  __shared__ float red[256];
  int tid = threadIdx.x;
  if (tid < PP) {
    float s = 0.f;
    for (int d = 0; d < HD; d++) { float v = protos[tid * HD + d]; s += v * v; }
    nrm[tid] = fmaxf(sqrtf(s), 1e-12f);
  }
  __syncthreads();
  for (int i = tid; i < PP * HD; i += 256) pn[i] = protos[i] / nrm[i / HD];
  __syncthreads();
  float part = 0.f;
  for (int t = tid; t < PP * PP * HD; t += 256) {
    int i = t / (PP * HD), r = t % (PP * HD), j = r / HD, d = r % HD;
    if (d > j) {
      float df = pn[j * HD + d] - pn[i * HD + d];
      float v = 1.f - df * df;
      if (v > 0.f) part += v;
    }
  }
  red[tid] = part; __syncthreads();
  for (int o = 128; o > 0; o >>= 1) { if (tid < o) red[tid] += red[tid + o]; __syncthreads(); }
  float dis = red[0] * 2.f / (float)(PP * (PP - 1)) / (float)HD;
  __syncthreads();
  float p2 = 0.f;
  for (int t = tid; t < (PP - 1) * HD; t += 256) {
    int p = t / HD, d = t % HD;
    float df = pn[(p + 1) * HD + d] - pn[p * HD + d];
    p2 += df * df;
  }
  red[tid] = p2; __syncthreads();
  for (int o = 128; o > 0; o >>= 1) { if (tid < o) red[tid] += red[tid + o]; __syncthreads(); }
  if (tid == 0) { outsc[1] = red[0] / (float)(PP - 1); outsc[2] = dis; }
}

// score softmax + l2norm(new_q) + per-row fea partial (one wave per row)
__global__ __launch_bounds__(256)
void score_kernel(const float* __restrict__ mha, const float* __restrict__ pn,
                  unsigned short* __restrict__ newqh, float* __restrict__ feapart) {
  int row = blockIdx.x * 8 + (threadIdx.x >> 5);
  if (row >= MQ) return;
  int lane = threadIdx.x & 31;
  const float* q = mha + (size_t)(NSUP + row) * HD;
  float sc[PP];
#pragma unroll
  for (int p = 0; p < PP; p++) {
    float s = 0.f;
    for (int d = lane; d < HD; d += 32) s += q[d] * pn[p * HD + d];
    for (int o = 16; o > 0; o >>= 1) s += __shfl_xor(s, o, 32);
    sc[p] = s;
  }
  float mx = sc[0]; int am = 0;
#pragma unroll
  for (int p = 1; p < PP; p++) if (sc[p] > mx) { mx = sc[p]; am = p; }
  float sum = 0.f;
#pragma unroll
  for (int p = 0; p < PP; p++) { sc[p] = __expf(sc[p] - mx); sum += sc[p]; }
  float inv = 1.f / sum;
  float vals[16];
  float ns = 0.f;
  for (int ii = 0, d = lane; d < HD; d += 32, ii++) {
    float v = 0.f;
#pragma unroll
    for (int p = 0; p < PP; p++) v += sc[p] * inv * pn[p * HD + d];
    vals[ii] = v; ns += v * v;
  }
  for (int o = 16; o > 0; o >>= 1) ns += __shfl_xor(ns, o, 32);
  float invn = 1.f / fmaxf(sqrtf(ns), 1e-12f);
  float fp = 0.f;
  for (int ii = 0, d = lane; d < HD; d += 32, ii++) {
    float v = vals[ii] * invn;
    newqh[(size_t)row * HD + d] = f32_to_bf16(v);
    float df = q[d] - pn[am * HD + d];
    fp += df * df;
  }
  for (int o = 16; o > 0; o >>= 1) fp += __shfl_xor(fp, o, 32);
  if (lane == 0) feapart[row] = fp;
}

__global__ __launch_bounds__(256)
void fea_reduce_kernel(const float* __restrict__ feapart, float* __restrict__ outsc) {
  __shared__ float red[256];
  float s = 0.f;
  for (int i = threadIdx.x; i < MQ; i += 256) s += feapart[i];
  red[threadIdx.x] = s; __syncthreads();
  for (int o = 128; o > 0; o >>= 1) { if (threadIdx.x < o) red[threadIdx.x] += red[threadIdx.x + o]; __syncthreads(); }
  if (threadIdx.x == 0) outsc[0] = red[0] / ((float)MQ * (float)HD);
}

// conv2 BN + relu + query residual, write NCHW output
__global__ __launch_bounds__(256)
void bn2_final_kernel(const float* __restrict__ Y2, const float* __restrict__ stats,
                      const float* __restrict__ g2, const float* __restrict__ b2,
                      const float* __restrict__ query, float* __restrict__ out) {
  long long idx = (long long)blockIdx.x * 256 + threadIdx.x;
  if (idx >= (long long)MQ * FD) return;
  int row = (int)(idx / FD), o = (int)(idx % FD);
  int n = row / S_HW, s = row % S_HW;
  float v = (Y2[idx] - stats[o]) * rsqrtf(stats[FD + o] + 1e-5f) * g2[o] + b2[o];
  v = v > 0.f ? v : 0.f;
  size_t oi = ((size_t)n * FD + o) * S_HW + s;
  out[oi] = v + query[oi];
}

// ---------------- host side ----------------
extern "C" void kernel_launch(void* const* d_in, const int* in_sizes, int n_in,
                              void* d_out, int out_size, void* d_ws, size_t ws_size,
                              hipStream_t stream) {
  (void)in_sizes; (void)n_in; (void)out_size; (void)ws_size;
  const float* key  = (const float*)d_in[0];
  const float* qry  = (const float*)d_in[1];
  const float* Wc   = (const float*)d_in[2];
  const float* bc   = (const float*)d_in[3];
  const float* g1   = (const float*)d_in[4];
  const float* b1   = (const float*)d_in[5];
  const float* Wq   = (const float*)d_in[6];
  const float* bq   = (const float*)d_in[7];
  const float* Wk   = (const float*)d_in[8];
  const float* bk   = (const float*)d_in[9];
  const float* Wv   = (const float*)d_in[10];
  const float* bv   = (const float*)d_in[11];
  const float* Wa   = (const float*)d_in[12];
  const float* ba   = (const float*)d_in[13];
  const float* lng  = (const float*)d_in[14];
  const float* lnb  = (const float*)d_in[15];
  const float* Wm   = (const float*)d_in[16];
  const float* Wo   = (const float*)d_in[17];
  const float* bo   = (const float*)d_in[18];
  const float* g2   = (const float*)d_in[19];
  const float* b2   = (const float*)d_in[20];
  float* out = (float*)d_out;
  char* ws = (char*)d_ws;

  auto AL = [](size_t x) { return (x + 255) & ~(size_t)255; };
  size_t off = 0;
  auto take = [&](size_t sz) { size_t o = off; off = AL(off + sz); return o; };

  size_t oWCH = take((size_t)HD * FD * 2);
  size_t oWQH = take((size_t)HD * HD * 2);
  size_t oWKH = take((size_t)HD * HD * 2);
  size_t oWVH = take((size_t)HD * HD * 2);
  size_t oWAH = take((size_t)HD * HD * 2);
  size_t oWOH = take((size_t)FD * HD * 2);
  size_t oA1  = take((size_t)M1 * FD * 2);
  size_t oY1  = take((size_t)M1 * HD * 4);
  size_t oST1 = take((size_t)2 * HD * 4);
  size_t oFE  = take((size_t)M1 * HD * 4);
  size_t oX   = take((size_t)MM * HD * 4);
  size_t oXH  = take((size_t)MM * HD * 2);
  size_t oQH  = take((size_t)MM * HD * 2);
  size_t oKH  = take((size_t)MM * HD * 2);
  size_t oVH  = take((size_t)MM * HD * 2);
  size_t oMHA = take((size_t)MM * HD * 4);
  size_t oPR  = take((size_t)PP * HD * 4);
  size_t oPN  = take((size_t)PP * HD * 4);
  size_t oFEA = take((size_t)MQ * 4);
  size_t oST2 = take((size_t)2 * FD * 4);
  // lifetime-based aliases
  size_t oY2  = oY1;   // Y1 dead after bn_relu
  size_t oCTX = oFE;   // feats dead after build_x
  size_t oCTXH = oXH;  // x_bf16 dead after qkv gemms
  size_t oNQH = oA1;   // A1 dead after conv1 gemm

  unsigned short* WcH = (unsigned short*)(ws + oWCH);
  unsigned short* WqH = (unsigned short*)(ws + oWQH);
  unsigned short* WkH = (unsigned short*)(ws + oWKH);
  unsigned short* WvH = (unsigned short*)(ws + oWVH);
  unsigned short* WaH = (unsigned short*)(ws + oWAH);
  unsigned short* WoH = (unsigned short*)(ws + oWOH);
  unsigned short* A1  = (unsigned short*)(ws + oA1);
  float* Y1    = (float*)(ws + oY1);
  float* ST1   = (float*)(ws + oST1);
  float* FEATS = (float*)(ws + oFE);
  float* X     = (float*)(ws + oX);
  unsigned short* XH  = (unsigned short*)(ws + oXH);
  unsigned short* QH  = (unsigned short*)(ws + oQH);
  unsigned short* KH  = (unsigned short*)(ws + oKH);
  unsigned short* VH  = (unsigned short*)(ws + oVH);
  float* CTX   = (float*)(ws + oCTX);
  unsigned short* CTXH = (unsigned short*)(ws + oCTXH);
  float* MHA   = (float*)(ws + oMHA);
  float* PR    = (float*)(ws + oPR);
  float* PN    = (float*)(ws + oPN);
  unsigned short* NQH = (unsigned short*)(ws + oNQH);
  float* FEAP  = (float*)(ws + oFEA);
  float* Y2    = (float*)(ws + oY2);
  float* ST2   = (float*)(ws + oST2);
  float* outsc = out + OUT_UPD;

  auto cdiv = [](long long a, long long b) { return (int)((a + b - 1) / b); };

  // weight conversions
  f2bf_kernel<<<cdiv(HD * FD, 256), 256, 0, stream>>>(Wc, WcH, HD * FD);
  f2bf_kernel<<<cdiv(HD * HD, 256), 256, 0, stream>>>(Wq, WqH, HD * HD);
  f2bf_kernel<<<cdiv(HD * HD, 256), 256, 0, stream>>>(Wk, WkH, HD * HD);
  f2bf_kernel<<<cdiv(HD * HD, 256), 256, 0, stream>>>(Wv, WvH, HD * HD);
  f2bf_kernel<<<cdiv(HD * HD, 256), 256, 0, stream>>>(Wa, WaH, HD * HD);
  f2bf_kernel<<<cdiv(FD * HD, 256), 256, 0, stream>>>(Wo, WoH, FD * HD);

  // dynamic LDS for the GEMM weight slab: 64 rows x K bf16
  const unsigned SM_K640 = 64 * FD * 2;   // 80 KB (conv1)
  const unsigned SM_K512 = 64 * HD * 2;   // 64 KB (others)

  // conv1: pack + gemm + bn + relu
  pack_key_kernel<<<cdiv((long long)M1 * FD, 256), 256, 0, stream>>>(key, A1);
  gemm_tn_wmma<0><<<dim3(HD / 64, cdiv(M1, 64)), 128, SM_K640, stream>>>(A1, WcH, bc, nullptr, Y1, nullptr, M1, HD, FD);
  bn_stats_kernel<<<HD, 256, 0, stream>>>(Y1, M1, HD, ST1);
  bn_relu_kernel<<<cdiv((long long)M1 * HD, 256), 256, 0, stream>>>(Y1, ST1, g1, b1, FEATS, M1, HD);

  // mha input (support mean + query feats)
  build_x_kernel<<<cdiv((long long)MM * HD, 256), 256, 0, stream>>>(FEATS, X, XH);

  // q,k,v projections (bf16 out)
  gemm_tn_wmma<1><<<dim3(HD / 64, MM / 64), 128, SM_K512, stream>>>(XH, WqH, bq, nullptr, nullptr, QH, MM, HD, HD);
  gemm_tn_wmma<1><<<dim3(HD / 64, MM / 64), 128, SM_K512, stream>>>(XH, WkH, bk, nullptr, nullptr, KH, MM, HD, HD);
  gemm_tn_wmma<1><<<dim3(HD / 64, MM / 64), 128, SM_K512, stream>>>(XH, WvH, bv, nullptr, nullptr, VH, MM, HD, HD);

  // attention (640 blocks, LDS-resident scores/att)
  attn_kernel<<<MB * NH, 256, ATT_SMEM, stream>>>(QH, KH, VH, CTX);

  // output projection + bias + residual, then layernorm in place
  f2bf_kernel<<<cdiv((long long)MM * HD, 256), 256, 0, stream>>>(CTX, CTXH, MM * HD);
  gemm_tn_wmma<2><<<dim3(HD / 64, MM / 64), 128, SM_K512, stream>>>(CTXH, WaH, ba, X, MHA, nullptr, MM, HD, HD);
  layernorm_kernel<<<cdiv(MM, 8), 256, 0, stream>>>(MHA, lng, lnb, MM);

  // prototypes + losses
  proto_kernel<<<1, 256, 0, stream>>>(MHA, Wm, PR);
  proto_loss_kernel<<<1, 256, 0, stream>>>(PR, PN, outsc);

  // query scoring / new_q / fea loss
  score_kernel<<<cdiv(MQ, 8), 256, 0, stream>>>(MHA, PN, NQH, FEAP);
  fea_reduce_kernel<<<1, 256, 0, stream>>>(FEAP, outsc);

  // conv2: gemm + bn + relu + residual -> d_out
  gemm_tn_wmma<0><<<dim3(FD / 64, cdiv(MQ, 64)), 128, SM_K512, stream>>>(NQH, WoH, bo, nullptr, Y2, nullptr, MQ, FD, HD);
  bn_stats_kernel<<<FD, 256, 0, stream>>>(Y2, MQ, FD, ST2);
  bn2_final_kernel<<<cdiv((long long)MQ * FD, 256), 256, 0, stream>>>(Y2, ST2, g2, b2, qry, out);
}
